// EncoderBlock_36438502540010
// MI455X (gfx1250) — compile-verified, matrix operand order checked by reading, not run on hardware
//
#include <hip/hip_runtime.h>

// ---------------- problem constants ----------------
#define EMB 384
#define NH  6
#define HD  64
#define FFD 1536
#define BSZ 256
#define TSZ 256
#define NTOK (BSZ * TSZ)   // 65536 tokens

// ---------------- vector types ----------------
typedef __attribute__((ext_vector_type(16))) __bf16          v16bf;
typedef __attribute__((ext_vector_type(8)))  float           v8f;
typedef __attribute__((ext_vector_type(8)))  unsigned short  ushort8;
typedef __attribute__((ext_vector_type(16))) unsigned short  ushort16;

// ---------------- bf16 helpers (RNE) ----------------
__device__ __forceinline__ unsigned short f2bf(float f) {
    unsigned int u = __builtin_bit_cast(unsigned int, f);
    u += 0x7FFFu + ((u >> 16) & 1u);
    return (unsigned short)(u >> 16);
}
__device__ __forceinline__ float bf2f(unsigned short s) {
    unsigned int u = ((unsigned int)s) << 16;
    return __builtin_bit_cast(float, u);
}

// ---------------- async global -> LDS copy (CDNA5 ASYNCcnt path) -------------
// 16B per lane: LDS[ldsOff] = MEM[gaddr]; tracked by ASYNCcnt.
__device__ __forceinline__ void async_copy16(const void* gaddr, void* ldsPtr) {
    unsigned lds = (unsigned)(size_t)ldsPtr;   // low 32 bits = LDS offset
    asm volatile("global_load_async_to_lds_b128 %0, %1, off"
                 :: "v"(lds), "v"(gaddr) : "memory");
}
__device__ __forceinline__ void async_wait0() {
    asm volatile("s_wait_asynccnt 0x0" ::: "memory");
}

// ---------------- WMMA fragment load ----------------
// 16-bit A-matrix 16x32 lane layout (CDNA5 ISA 7.12.2):
//   lane l: row m = l&15, kb = (l>>4)*8, elements k = kb..kb+7, kb+16..kb+23.
// B operand is loaded from Bt (row-major [N][K]) with the identical routine.
__device__ __forceinline__ v16bf make_frag(ushort8 lo, ushort8 hi) {
    ushort16 c = __builtin_shufflevector(lo, hi,
        0,1,2,3,4,5,6,7,8,9,10,11,12,13,14,15);
    return __builtin_bit_cast(v16bf, c);
}
// tile: pointer to element [row0][col0] of an LDS tile with row stride `stride` (ushorts)
__device__ __forceinline__ v16bf frag_ld(const unsigned short* tile, int stride, int lane) {
    const unsigned short* p = tile + (lane & 15) * stride + ((lane >> 4) << 3);
    ushort8 lo = *(const ushort8*)(p);
    ushort8 hi = *(const ushort8*)(p + 16);
    return make_frag(lo, hi);
}

__device__ __forceinline__ v8f wmma_bf16(v16bf a, v16bf b, v8f c) {
    return __builtin_amdgcn_wmma_f32_16x16x32_bf16(
        /*neg_a=*/false, a, /*neg_b=*/false, b,
        /*c_mod=*/(short)0, c, /*reuse_a=*/false, /*reuse_b=*/false);
}

// ---------------- generic bf16 GEMM: C[M,N] = A[M,K] * Bt[N,K]^T ----------------
// Block tile 128x128, K-step 32, 256 threads = 8 waves (2x4), wave tile 64x32.
// Double-buffered LDS, async global->LDS staging, one barrier per K-step.
template <typename Epi>
__global__ __launch_bounds__(256) void gemm_bf16_nt(
    const unsigned short* __restrict__ A,
    const unsigned short* __restrict__ Bt,
    int M, int N, int K, Epi epi)
{
    __shared__ unsigned short As[2][128][40];   // 32 + 8 pad (keeps 16B alignment)
    __shared__ unsigned short Bs[2][128][40];

    const int tid  = threadIdx.x;
    const int lane = tid & 31;
    const int wid  = tid >> 5;
    const int tiles_n = N >> 7;
    const int bm = (int)blockIdx.x / tiles_n;
    const int bn = (int)blockIdx.x % tiles_n;
    const int m0 = bm << 7, n0 = bn << 7;
    const int wm = (wid & 1) * 64;     // wave M offset in tile
    const int wn = (wid >> 1) * 32;    // wave N offset in tile

    // per-thread staging coordinates: 512 16B-chunks per tile, 2 per thread
    const int r0  = tid >> 2,          cc0 = (tid & 3) << 3;
    const int r1  = (tid + 256) >> 2,  cc1 = ((tid + 256) & 3) << 3;

    auto stage = [&](int bsel, int kk) {
        async_copy16(&A[(size_t)(m0 + r0) * K + kk + cc0], &As[bsel][r0][cc0]);
        async_copy16(&Bt[(size_t)(n0 + r0) * K + kk + cc0], &Bs[bsel][r0][cc0]);
        async_copy16(&A[(size_t)(m0 + r1) * K + kk + cc1], &As[bsel][r1][cc1]);
        async_copy16(&Bt[(size_t)(n0 + r1) * K + kk + cc1], &Bs[bsel][r1][cc1]);
    };

    v8f acc[4][2] = {};

    stage(0, 0);
    async_wait0();
    __syncthreads();

    int buf = 0;
    for (int k0 = 0; k0 < K; k0 += 32, buf ^= 1) {
        const bool more = (k0 + 32) < K;
        if (more) stage(buf ^ 1, k0 + 32);   // prefetch next tile into other buffer

        v16bf af[4], bfv[2];
        #pragma unroll
        for (int i = 0; i < 4; ++i) af[i]  = frag_ld(&As[buf][wm + i * 16][0], 40, lane);
        #pragma unroll
        for (int j = 0; j < 2; ++j) bfv[j] = frag_ld(&Bs[buf][wn + j * 16][0], 40, lane);
        #pragma unroll
        for (int i = 0; i < 4; ++i)
            #pragma unroll
            for (int j = 0; j < 2; ++j)
                acc[i][j] = wmma_bf16(af[i], bfv[j], acc[i][j]);

        if (more) async_wait0();
        __syncthreads();
    }

    // C/D layout: lane l -> n = l&15; VGPR r -> m = r + (l>>4)*8
    #pragma unroll
    for (int i = 0; i < 4; ++i)
        #pragma unroll
        for (int j = 0; j < 2; ++j)
            #pragma unroll
            for (int r = 0; r < 8; ++r) {
                int m = m0 + wm + i * 16 + ((lane >> 4) << 3) + r;
                int n = n0 + wn + j * 16 + (lane & 15);
                epi(m, n, acc[i][j][r]);
            }
}

// ---------------- GEMM epilogues ----------------
struct QkvEpi {
    unsigned short *q, *k, *vt;
    __device__ void operator()(int m, int n, float v) const {
        int b = m >> 8, t = m & 255;
        if (n < EMB) {                       // Q (pre-scaled by D^-0.5 = 0.125)
            int h = n >> 6, d = n & 63;
            q[(((size_t)(b * NH + h) * TSZ + t) << 6) + d] = f2bf(v * 0.125f);
        } else if (n < 2 * EMB) {            // K  [b,h,t,d]
            int nn = n - EMB; int h = nn >> 6, d = nn & 63;
            k[(((size_t)(b * NH + h) * TSZ + t) << 6) + d] = f2bf(v);
        } else {                             // V transposed [b,h,d,t]
            int nn = n - 2 * EMB; int h = nn >> 6, d = nn & 63;
            vt[(((size_t)(b * NH + h) * HD + d) << 8) + t] = f2bf(v);
        }
    }
};
struct WoEpi {
    const float *x, *bo; float *x1;
    __device__ void operator()(int m, int n, float v) const {
        size_t i = (size_t)m * EMB + n;
        x1[i] = v + x[i] + bo[n];
    }
};
struct Ffn1Epi {
    const float* b1; unsigned short* f;
    __device__ void operator()(int m, int n, float v) const {
        float r = v + b1[n];
        f[(size_t)m * FFD + n] = f2bf(r > 0.f ? r : 0.f);
    }
};
struct Ffn2Epi {
    const float *x1, *b2; float* out;
    __device__ void operator()(int m, int n, float v) const {
        size_t i = (size_t)m * EMB + n;
        out[i] = v + x1[i] + b2[n];
    }
};

// ---------------- LayerNorm: one wave per 384-elem row, bf16 out ----------------
__global__ __launch_bounds__(256) void layernorm_to_bf16(
    const float* __restrict__ x, const float* __restrict__ g,
    const float* __restrict__ b, unsigned short* __restrict__ out)
{
    const int lane = threadIdx.x & 31;
    const int row  = blockIdx.x * 8 + (threadIdx.x >> 5);
    const float* xr = x + (size_t)row * EMB;
    float v[12]; float s = 0.f;
    #pragma unroll
    for (int i = 0; i < 12; ++i) { v[i] = xr[lane + i * 32]; s += v[i]; }
    #pragma unroll
    for (int o = 16; o > 0; o >>= 1) s += __shfl_xor(s, o, 32);
    const float mu = s * (1.0f / EMB);
    float qv = 0.f;
    #pragma unroll
    for (int i = 0; i < 12; ++i) { float d = v[i] - mu; qv += d * d; }
    #pragma unroll
    for (int o = 16; o > 0; o >>= 1) qv += __shfl_xor(qv, o, 32);
    const float rs = rsqrtf(qv * (1.0f / EMB) + 1e-5f);
    unsigned short* orow = out + (size_t)row * EMB;
    #pragma unroll
    for (int i = 0; i < 12; ++i) {
        int c = lane + i * 32;
        orow[c] = f2bf((v[i] - mu) * rs * g[c] + b[c]);
    }
}

// ---------------- weight transpose + fp32->bf16: dst[N][K] = src[K][N]^T ----------
__global__ void transpose_f32_to_bf16(const float* __restrict__ src,
                                      unsigned short* __restrict__ dst,
                                      int Kd, int Nd)
{
    int idx = blockIdx.x * 256 + threadIdx.x;
    if (idx >= Kd * Nd) return;
    int o = idx / Kd, c = idx - o * Kd;
    dst[idx] = f2bf(src[(size_t)c * Nd + o]);
}

// ---------------- attention: one WG per (b, h, 64-query block) ----------------
__global__ __launch_bounds__(256) void attn_kernel(
    const unsigned short* __restrict__ q, const unsigned short* __restrict__ k,
    const unsigned short* __restrict__ vt, unsigned short* __restrict__ o)
{
    __shared__ unsigned short Qs [64][72];     //  9 216 B
    __shared__ unsigned short Ks [256][72];    // 36 864 B
    __shared__ unsigned short Vts[64][264];    // 33 792 B
    __shared__ unsigned short Ps [64][264];    // 33 792 B  (scores -> probs in place)

    const int tid = threadIdx.x, lane = tid & 31, wid = tid >> 5;
    const int qblk = blockIdx.x & 3;
    const int bh   = blockIdx.x >> 2;          // b*NH + h
    const int h = bh % NH, b = bh / NH;

    const size_t qbase = ((size_t)bh * TSZ + qblk * 64) * HD;
    const size_t kbase = (size_t)bh * TSZ * HD;
    const size_t vbase = (size_t)bh * HD * TSZ;

    // async-stage Q (64x64), K (256x64), V^T (64x256) into LDS
    #pragma unroll
    for (int s = 0; s < 2; ++s) {
        int c = tid + s * 256; int r = c >> 3, cc = (c & 7) << 3;
        async_copy16(&q[qbase + (size_t)r * HD + cc], &Qs[r][cc]);
    }
    #pragma unroll
    for (int s = 0; s < 8; ++s) {
        int c = tid + s * 256; int r = c >> 3, cc = (c & 7) << 3;
        async_copy16(&k[kbase + (size_t)r * HD + cc], &Ks[r][cc]);
    }
    #pragma unroll
    for (int s = 0; s < 8; ++s) {
        int c = tid + s * 256; int r = c >> 5, cc = (c & 31) << 3;
        async_copy16(&vt[vbase + (size_t)r * TSZ + cc], &Vts[r][cc]);
    }
    async_wait0();
    __syncthreads();

    // S = Q * K^T  (64x256); waves 2x4 -> each 32x64
    {
        const int wm = (wid & 1) * 32, wn = (wid >> 1) * 64;
        v8f acc[2][4] = {};
        #pragma unroll
        for (int kk = 0; kk < HD; kk += 32) {
            v16bf af[2], bfv[4];
            #pragma unroll
            for (int i = 0; i < 2; ++i) af[i]  = frag_ld(&Qs[wm + i * 16][kk], 72, lane);
            #pragma unroll
            for (int j = 0; j < 4; ++j) bfv[j] = frag_ld(&Ks[wn + j * 16][kk], 72, lane);
            #pragma unroll
            for (int i = 0; i < 2; ++i)
                #pragma unroll
                for (int j = 0; j < 4; ++j)
                    acc[i][j] = wmma_bf16(af[i], bfv[j], acc[i][j]);
        }
        #pragma unroll
        for (int i = 0; i < 2; ++i)
            #pragma unroll
            for (int j = 0; j < 4; ++j)
                #pragma unroll
                for (int r = 0; r < 8; ++r) {
                    int m = wm + i * 16 + ((lane >> 4) << 3) + r;
                    int n = wn + j * 16 + (lane & 15);
                    Ps[m][n] = f2bf(acc[i][j][r]);
                }
    }
    __syncthreads();

    // row softmax: 4 threads per row, 64 columns each, quad reduction via shfl
    {
        const int row = tid >> 2;
        const int c0  = (tid & 3) * 64;
        float mx = -3.402823e38f;
        for (int s = 0; s < 64; ++s) mx = fmaxf(mx, bf2f(Ps[row][c0 + s]));
        mx = fmaxf(mx, __shfl_xor(mx, 1, 32));
        mx = fmaxf(mx, __shfl_xor(mx, 2, 32));
        float sum = 0.f;
        for (int s = 0; s < 64; ++s) {
            float e = __expf(bf2f(Ps[row][c0 + s]) - mx);
            sum += e;
            Ps[row][c0 + s] = f2bf(e);
        }
        sum += __shfl_xor(sum, 1, 32);
        sum += __shfl_xor(sum, 2, 32);
        const float inv = 1.0f / sum;
        for (int s = 0; s < 64; ++s)
            Ps[row][c0 + s] = f2bf(bf2f(Ps[row][c0 + s]) * inv);
    }
    __syncthreads();

    // O = P * V  (64x64); waves 2x4 -> each 32 rows x 16 cols
    {
        const int wm = (wid & 1) * 32;
        const int wn = (wid >> 1) * 16;
        v8f acc0 = {}, acc1 = {};
        #pragma unroll
        for (int s0 = 0; s0 < TSZ; s0 += 32) {
            v16bf bfv = frag_ld(&Vts[wn][s0], 264, lane);
            v16bf a0  = frag_ld(&Ps[wm][s0],      264, lane);
            v16bf a1  = frag_ld(&Ps[wm + 16][s0], 264, lane);
            acc0 = wmma_bf16(a0, bfv, acc0);
            acc1 = wmma_bf16(a1, bfv, acc1);
        }
        const size_t row0 = (size_t)b * TSZ + qblk * 64;
        const int d = wn + (lane & 15);
        #pragma unroll
        for (int r = 0; r < 8; ++r) {
            int m = wm + ((lane >> 4) << 3) + r;
            o[(row0 + m)      * EMB + h * HD + d] = f2bf(acc0[r]);
            o[(row0 + m + 16) * EMB + h * HD + d] = f2bf(acc1[r]);
        }
    }
}

// ---------------- host-side orchestration ----------------
extern "C" void kernel_launch(void* const* d_in, const int* in_sizes, int n_in,
                              void* d_out, int out_size, void* d_ws, size_t ws_size,
                              hipStream_t stream) {
    const float* x   = (const float*)d_in[0];
    const float* Wq  = (const float*)d_in[1];
    const float* Wk  = (const float*)d_in[2];
    const float* Wv  = (const float*)d_in[3];
    const float* Wo  = (const float*)d_in[4];
    const float* bo  = (const float*)d_in[5];
    const float* W1  = (const float*)d_in[6];
    const float* b1  = (const float*)d_in[7];
    const float* W2  = (const float*)d_in[8];
    const float* b2  = (const float*)d_in[9];
    const float* g1  = (const float*)d_in[10];
    const float* be1 = (const float*)d_in[11];
    const float* g2  = (const float*)d_in[12];
    const float* be2 = (const float*)d_in[13];
    float* out = (float*)d_out;

    // workspace layout (bytes, all 256-aligned)
    char* ws = (char*)d_ws;
    unsigned short* wqkv_t = (unsigned short*)(ws + 0);           // 1152x384 bf16
    unsigned short* wo_t   = (unsigned short*)(ws + 884736);      //  384x384
    unsigned short* w1_t   = (unsigned short*)(ws + 1179648);     // 1536x384
    unsigned short* w2_t   = (unsigned short*)(ws + 2359296);     //  384x1536
    unsigned short* hbuf   = (unsigned short*)(ws + 3538944);     // 65536x384 bf16 (h1, later h2)
    unsigned short* qb     = (unsigned short*)(ws + 53870592);    // q  [B,H,T,D] bf16
    unsigned short* kb     = qb  + (size_t)NTOK * EMB;            // k  [B,H,T,D]
    unsigned short* vtb    = kb  + (size_t)NTOK * EMB;            // v^T[B,H,D,T]
    unsigned short* ob     = vtb + (size_t)NTOK * EMB;            // attn out [B*T, 384]
    unsigned short* fbuf   = qb;                                  // FFN hidden aliases q/k/vt/o (4*384 = 1536)
    float* x1              = (float*)(ws + 255197184);            // 65536x384 fp32

    // weight prep: transpose + bf16
    const int n384 = EMB * EMB, nff = EMB * FFD;
    transpose_f32_to_bf16<<<(n384 + 255) / 256, 256, 0, stream>>>(Wq, wqkv_t,            EMB, EMB);
    transpose_f32_to_bf16<<<(n384 + 255) / 256, 256, 0, stream>>>(Wk, wqkv_t + n384,     EMB, EMB);
    transpose_f32_to_bf16<<<(n384 + 255) / 256, 256, 0, stream>>>(Wv, wqkv_t + 2 * n384, EMB, EMB);
    transpose_f32_to_bf16<<<(n384 + 255) / 256, 256, 0, stream>>>(Wo, wo_t,              EMB, EMB);
    transpose_f32_to_bf16<<<(nff  + 255) / 256, 256, 0, stream>>>(W1, w1_t,              EMB, FFD);
    transpose_f32_to_bf16<<<(nff  + 255) / 256, 256, 0, stream>>>(W2, w2_t,              FFD, EMB);

    // h1 = LN(x)
    layernorm_to_bf16<<<NTOK / 8, 256, 0, stream>>>(x, g1, be1, hbuf);

    // q,k,v = h1 @ [Wq|Wk|Wv]
    gemm_bf16_nt<QkvEpi><<<(NTOK / 128) * (3 * EMB / 128), 256, 0, stream>>>(
        hbuf, wqkv_t, NTOK, 3 * EMB, EMB, QkvEpi{qb, kb, vtb});

    // attention
    attn_kernel<<<BSZ * NH * 4, 256, 0, stream>>>(qb, kb, vtb, ob);

    // x1 = x + o @ Wo + bo
    gemm_bf16_nt<WoEpi><<<(NTOK / 128) * (EMB / 128), 256, 0, stream>>>(
        ob, wo_t, NTOK, EMB, EMB, WoEpi{x, bo, x1});

    // h2 = LN(x1)
    layernorm_to_bf16<<<NTOK / 8, 256, 0, stream>>>(x1, g2, be2, hbuf);

    // f = relu(h2 @ W1 + b1)
    gemm_bf16_nt<Ffn1Epi><<<(NTOK / 128) * (FFD / 128), 256, 0, stream>>>(
        hbuf, w1_t, NTOK, FFD, EMB, Ffn1Epi{b1, fbuf});

    // out = x1 + f @ W2 + b2
    gemm_bf16_nt<Ffn2Epi><<<(NTOK / 128) * (EMB / 128), 256, 0, stream>>>(
        fbuf, w2_t, NTOK, EMB, FFD, Ffn2Epi{x1, b2, out});
}
